// EnhancedGatedMDTA_38946763440645
// MI455X (gfx1250) — matrix operand order repaired and checked by reading, not compiled, not run on hardware
//
#include <hip/hip_runtime.h>
#include <hip/hip_bf16.h>
#include <math.h>

#define NPIX   16384
#define HPIX   128
#define CDIM   192
#define C4     768
#define BATCH  8
#define NHEADS 4
#define HD     48

typedef __bf16 bf16_t;
typedef __attribute__((ext_vector_type(8)))  __bf16 v8bf;
typedef __attribute__((ext_vector_type(16))) __bf16 v16bf;
typedef __attribute__((ext_vector_type(8)))  float  v8f;
typedef __attribute__((ext_vector_type(4)))  unsigned int u32x4;
typedef __attribute__((ext_vector_type(8)))  int    i32x8;
typedef __attribute__((ext_vector_type(4)))  int    i32x4;

#if defined(__has_builtin)
#if __has_builtin(__builtin_amdgcn_tensor_load_to_lds) && \
    __has_builtin(__builtin_amdgcn_s_wait_tensorcnt)
#define HAVE_TDM 1
#endif
#endif
#if __has_include(<hip/amd_detail/amd_gfx1250_TDM.h>)
#define TDM_6ARG 1
#else
#define TDM_6ARG 0
#endif

// ---------------------------------------------------------------- WMMA helpers
__device__ __forceinline__ v16bf cat16(v8bf lo, v8bf hi) {
  return __builtin_shufflevector(lo, hi, 0,1,2,3,4,5,6,7,8,9,10,11,12,13,14,15);
}
// A fragment (16x32 bf16, MxK): lane holds row m = lane%16, half = lane/16.
// element e -> K = (e<8 ? e : e+8) + 8*half  (two contiguous 16B runs)
__device__ __forceinline__ v16bf load_a_frag(const bf16_t* row, int half) {
  v8bf lo = *(const v8bf*)(row + 8 * half);
  v8bf hi = *(const v8bf*)(row + 16 + 8 * half);
  return cat16(lo, hi);
}
// B fragment (32x16 bf16, KxN) from an n-major row: lane n = lane%16,
// khalf = lane/16; element e -> K = (e<8 ? e : e+8) + 8*khalf (two contiguous 16B runs)
__device__ __forceinline__ v16bf load_b_frag(const bf16_t* row, int khalf) {
  v8bf lo = *(const v8bf*)(row + 8 * khalf);
  v8bf hi = *(const v8bf*)(row + 16 + 8 * khalf);
  return cat16(lo, hi);
}
__device__ __forceinline__ v8f wmma_bf16(v16bf a, v16bf b, v8f c) {
  return __builtin_amdgcn_wmma_f32_16x16x32_bf16(false, a, false, b, (short)0, c,
                                                 false, false);
}
__device__ __forceinline__ float sigmoidf_(float x) { return 1.0f / (1.0f + expf(-x)); }

// ------------------------------------------------- K1: per-channel avg/max pool
__global__ void pool_kernel(const float* __restrict__ x, float* __restrict__ pool) {
  int bc = blockIdx.x;                 // b*192 + c
  int b = bc / CDIM, c = bc % CDIM;
  const float* p = x + (size_t)bc * NPIX;
  float s = 0.f, m = -3.4e38f;
  for (int i = threadIdx.x * 4; i < NPIX; i += 256 * 4) {
#pragma unroll
    for (int j = 0; j < 4; ++j) {
      float v = p[i + j];
      s += v;
      m = fmaxf(m, v);
    }
  }
  __shared__ float ss[256], sm[256];
  ss[threadIdx.x] = s; sm[threadIdx.x] = m;
  __syncthreads();
  for (int w = 128; w > 0; w >>= 1) {
    if (threadIdx.x < w) {
      ss[threadIdx.x] += ss[threadIdx.x + w];
      sm[threadIdx.x] = fmaxf(sm[threadIdx.x], sm[threadIdx.x + w]);
    }
    __syncthreads();
  }
  if (threadIdx.x == 0) {
    pool[b * 2 * CDIM + c] = ss[0] * (1.0f / NPIX);   // avg block
    pool[b * 2 * CDIM + CDIM + c] = sm[0];            // max block
  }
}

// ------------------------------------------------- K2: channel gate (fc1+fc2)
__global__ void gate_kernel(const float* __restrict__ pool,
                            const float* __restrict__ wfc1,
                            const float* __restrict__ wfc2,
                            float* __restrict__ cg) {
  int b = blockIdx.x;
  __shared__ float pl[2 * CDIM];
  __shared__ float sq[HD];
  for (int i = threadIdx.x; i < 2 * CDIM; i += blockDim.x)
    pl[i] = pool[b * 2 * CDIM + i];
  __syncthreads();
  if (threadIdx.x < HD) {
    float s = 0.f;
    for (int i = 0; i < 2 * CDIM; ++i) s += wfc1[threadIdx.x * 2 * CDIM + i] * pl[i];
    sq[threadIdx.x] = fmaxf(s, 0.f);
  }
  __syncthreads();
  if (threadIdx.x < CDIM) {
    float s = 0.f;
    for (int j = 0; j < HD; ++j) s += wfc2[threadIdx.x * HD + j] * sq[j];
    cg[b * CDIM + threadIdx.x] = sigmoidf_(s);
  }
}

// ----------------------------------------- K3: per-pixel channel mean/max stats
__global__ void spstat_kernel(const float* __restrict__ x, float* __restrict__ spstat) {
  size_t flat = (size_t)blockIdx.x * 256 + threadIdx.x;   // b*NPIX + n
  int n = (int)(flat & (NPIX - 1));
  int b = (int)(flat >> 14);
  const float* p = x + (size_t)b * CDIM * NPIX + n;
  float s = 0.f, m = -3.4e38f;
  for (int c = 0; c < CDIM; ++c) {
    float v = p[(size_t)c * NPIX];
    s += v;
    m = fmaxf(m, v);
  }
  spstat[((size_t)b * 2 + 0) * NPIX + n] = s * (1.0f / CDIM);
  spstat[((size_t)b * 2 + 1) * NPIX + n] = m;
}

// ------------------------------------------------- K4: 7x7 spatial gate conv
__global__ void spconv_kernel(const float* __restrict__ spstat,
                              const float* __restrict__ wsp,
                              float* __restrict__ sg) {
  size_t flat = (size_t)blockIdx.x * 256 + threadIdx.x;
  int n = (int)(flat & (NPIX - 1));
  int b = (int)(flat >> 14);
  int y = n >> 7, x = n & 127;
  float s = 0.f;
  for (int ci = 0; ci < 2; ++ci) {
    const float* src = spstat + ((size_t)b * 2 + ci) * NPIX;
    const float* wk = wsp + ci * 49;
    for (int ky = 0; ky < 7; ++ky) {
      int yy = y + ky - 3;
      if (yy < 0 || yy >= HPIX) continue;
      for (int kx = 0; kx < 7; ++kx) {
        int xx = x + kx - 3;
        if (xx < 0 || xx >= HPIX) continue;
        s += src[yy * HPIX + xx] * wk[ky * 7 + kx];
      }
    }
  }
  sg[(size_t)b * NPIX + n] = sigmoidf_(s);
}

// ------------------------------------------------- K5: apply gates -> bf16 x (x8 wide)
__global__ void applygates_kernel(const float* __restrict__ x,
                                  const float* __restrict__ cg,
                                  const float* __restrict__ sg,
                                  bf16_t* __restrict__ xg) {
  size_t e0 = ((size_t)blockIdx.x * 256 + threadIdx.x) * 8;  // (b*C + c)*N + n
  int n = (int)(e0 & (NPIX - 1));
  size_t bc = e0 >> 14;
  int b = (int)(bc / CDIM);
  float c = cg[bc];
  const float* xp = x + e0;
  const float* sp = sg + (size_t)b * NPIX + n;
  v8bf r;
#pragma unroll
  for (int j = 0; j < 8; ++j) r[j] = (bf16_t)(xp[j] * c * sp[j]);
  *(v8bf*)(xg + e0) = r;
}

// ------------------------------------------------- f32 -> bf16 weight convert
__global__ void cvt_bf16_kernel(const float* __restrict__ src, bf16_t* __restrict__ dst,
                                int count) {
  int i = blockIdx.x * 256 + threadIdx.x;
  if (i < count) dst[i] = (bf16_t)src[i];
}

// ------------------------------------------------- generic bf16 WMMA GEMM
// Y[b][m][n] = sum_k W[m][k] * X[b][k][n],  N fixed = NPIX, K multiple of 32.
// Block tile: 64 (M) x 128 (N); 8 waves, each wave: 16 (M) x 64 (N) -> 4 C tiles.
// A-tile (row-major 64x32 copy) staged by the Tensor Data Mover when available.
template <bool OUT_BF16>
__global__ void gemm_kernel(const bf16_t* __restrict__ W, const bf16_t* __restrict__ X,
                            void* __restrict__ Y, int M, int K, long xbstride,
                            long ybstride) {
  __shared__ alignas(16) bf16_t Atile[64][32];    // m-major
  __shared__ alignas(16) bf16_t Btile[128][32];   // n-major (transposed stage)
  const bf16_t* Xb = X + (size_t)blockIdx.z * xbstride;
  int m0 = blockIdx.y * 64, n0 = blockIdx.x * 128;
  int tid = threadIdx.x, lane = tid & 31, wid = tid >> 5;
  int wm = wid & 3, wn = wid >> 2;
  v8f acc[4] = {};
  for (int kk = 0; kk < K; kk += 32) {
#ifdef HAVE_TDM
    if (wid == 0) {
      // Tensor DMA: 2D tile (tile_dim0=32 K-contig, tile_dim1=64 rows) from W.
      unsigned long long ga =
          (unsigned long long)(const void*)(W + (size_t)m0 * K + kk);
      unsigned int lds = (unsigned int)(size_t)(&Atile[0][0]);
      u32x4 g0;
      i32x8 g1;
      i32x4 gz;
      gz[0] = 0; gz[1] = 0; gz[2] = 0; gz[3] = 0;
      g0[0] = 1u;                                           // count=1, user D#
      g0[1] = lds;                                          // lds_addr
      g0[2] = (unsigned int)ga;                             // global_addr lo
      g0[3] = (unsigned int)((ga >> 32) & 0x01FFFFFFu) | (2u << 30);  // type=2
      g1[0] = 0x00010000;                  // workgroup_mask=0, data_size=1 (2B)
      g1[1] = (int)(((unsigned)K & 0xFFFFu) << 16);   // tensor_dim0 (low16)
      g1[2] = (int)(((unsigned)M & 0xFFFFu) << 16);   // tensor_dim1 (low16)
      g1[3] = (int)(32u << 16);            // tile_dim0 = 32
      g1[4] = 64;                          // tile_dim1 = 64, tile_dim2 = 0
      g1[5] = K;                           // tensor_dim0_stride (low32)
      g1[6] = 0;
      g1[7] = 0;
#if TDM_6ARG
      i32x8 g8;
      g8[0]=0; g8[1]=0; g8[2]=0; g8[3]=0; g8[4]=0; g8[5]=0; g8[6]=0; g8[7]=0;
      __builtin_amdgcn_tensor_load_to_lds(g0, g1, gz, gz, g8, 0);
#else
      __builtin_amdgcn_tensor_load_to_lds(g0, g1, gz, gz, 0);
#endif
      __builtin_amdgcn_s_wait_tensorcnt(0);
    }
#else
    for (int c = tid; c < 64 * 32 / 8; c += 256) {   // 128-bit row-major copy
      int row = c >> 2, c8 = c & 3;
      *(v8bf*)(&Atile[row][c8 * 8]) =
          *(const v8bf*)(W + (size_t)(m0 + row) * K + kk + c8 * 8);
    }
#endif
    for (int c = tid; c < 32 * 128 / 8; c += 256) {  // wide read, transposed store
      int row = c >> 4, c8 = c & 15;
      v8bf d = *(const v8bf*)(Xb + (size_t)(kk + row) * NPIX + n0 + c8 * 8);
#pragma unroll
      for (int j = 0; j < 8; ++j) Btile[c8 * 8 + j][row] = d[j];
    }
    __syncthreads();
    if (kk + 32 < K)
      __builtin_prefetch(Xb + (size_t)(kk + 32) * NPIX + n0, 0, 1);
    v16bf af = load_a_frag(&Atile[wm * 16 + (lane & 15)][0], lane >> 4);
#pragma unroll
    for (int t = 0; t < 4; ++t) {
      v16bf bf = load_b_frag(&Btile[wn * 64 + t * 16 + (lane & 15)][0], lane >> 4);
      acc[t] = wmma_bf16(af, bf, acc[t]);
    }
    __syncthreads();
  }
#pragma unroll
  for (int t = 0; t < 4; ++t)
#pragma unroll
    for (int r = 0; r < 8; ++r) {
      int mm = m0 + wm * 16 + r + 8 * (lane >> 4);
      int nn = n0 + wn * 64 + t * 16 + (lane & 15);
      float v = acc[t][r];
      if constexpr (OUT_BF16)
        (((bf16_t*)Y) + (size_t)blockIdx.z * ybstride)[(size_t)mm * NPIX + nn] =
            (bf16_t)v;
      else
        (((float*)Y) + (size_t)blockIdx.z * ybstride)[(size_t)mm * NPIX + nn] = v;
    }
}

// ------------------------------------------------- K7: depthwise 3x3 (bf16)
__global__ void dwconv_kernel(const bf16_t* __restrict__ qkv,
                              const float* __restrict__ wdw,
                              bf16_t* __restrict__ out) {
  size_t flat = (size_t)blockIdx.x * 256 + threadIdx.x;  // (b*C4 + ch)*N + n
  int n = (int)(flat & (NPIX - 1));
  size_t bc = flat >> 14;
  int ch = (int)(bc % C4);
  int y = n >> 7, x = n & 127;
  const bf16_t* in = qkv + (bc << 14);
  const float* wk = wdw + ch * 9;
  float s = 0.f;
#pragma unroll
  for (int dy = -1; dy <= 1; ++dy) {
    int yy = y + dy;
    if (yy < 0 || yy >= HPIX) continue;
#pragma unroll
    for (int dx = -1; dx <= 1; ++dx) {
      int xx = x + dx;
      if (xx < 0 || xx >= HPIX) continue;
      s += (float)in[yy * HPIX + xx] * wk[(dy + 1) * 3 + (dx + 1)];
    }
  }
  out[flat] = (bf16_t)s;
}

// ------------------------------------------------- K8: 1/L2-norm for q,k rows
__global__ void norm_kernel(const bf16_t* __restrict__ qkvdw, float* __restrict__ rnorm) {
  int bi = blockIdx.x;                  // b*384 + idx (idx<192 -> q, else k)
  int b = bi / (2 * CDIM), idx = bi % (2 * CDIM);
  int ch = (idx < CDIM) ? idx : (2 * CDIM + (idx - CDIM));
  const bf16_t* p = qkvdw + ((size_t)b * C4 + ch) * NPIX;
  float s = 0.f;
  for (int i = threadIdx.x * 8; i < NPIX; i += 256 * 8) {
    v8bf d = *(const v8bf*)(p + i);
#pragma unroll
    for (int j = 0; j < 8; ++j) {
      float v = (float)d[j];
      s += v * v;
    }
  }
  __shared__ float ss[256];
  ss[threadIdx.x] = s;
  __syncthreads();
  for (int w = 128; w > 0; w >>= 1) {
    if (threadIdx.x < w) ss[threadIdx.x] += ss[threadIdx.x + w];
    __syncthreads();
  }
  if (threadIdx.x == 0) rnorm[bi] = 1.0f / fmaxf(sqrtf(ss[0]), 1e-12f);
}

// ------------------------------------------------- K9: attn = softmax(q k^T * T)
// L2-normalization is separable: attn_raw[m][n] scaled by rq[m]*rk[n] afterwards.
// All 8 waves accumulate disjoint K slices with fragments loaded straight from
// global (two b128 loads each), reduced across waves via ds_add_f32 atomics.
__global__ void attn_kernel(const bf16_t* __restrict__ qkvdw,
                            const float* __restrict__ rnorm,
                            const float* __restrict__ temp,
                            float* __restrict__ attnG) {
  int head = blockIdx.x, b = blockIdx.y;
  __shared__ float attnS[HD][64];
  int tid = threadIdx.x, lane = tid & 31, wid = tid >> 5;
  for (int i = tid; i < HD * 64; i += 256) ((float*)attnS)[i] = 0.f;
  __syncthreads();
  const bf16_t* qbase = qkvdw + ((size_t)b * C4 + head * HD) * NPIX;
  const bf16_t* kbase = qkvdw + ((size_t)b * C4 + 2 * CDIM + head * HD) * NPIX;
  v8f acc[3][3] = {};
  for (int it = 0; it < NPIX / (8 * 32); ++it) {   // 64 iterations, no barriers
    int k0 = (it * 8 + wid) * 32;
    v16bf af[3], bf[3];
#pragma unroll
    for (int mt = 0; mt < 3; ++mt)
      af[mt] = load_a_frag(qbase + (size_t)(mt * 16 + (lane & 15)) * NPIX + k0,
                           lane >> 4);
#pragma unroll
    for (int nt = 0; nt < 3; ++nt)
      bf[nt] = load_b_frag(kbase + (size_t)(nt * 16 + (lane & 15)) * NPIX + k0,
                           lane >> 4);
#pragma unroll
    for (int mt = 0; mt < 3; ++mt)
#pragma unroll
      for (int nt = 0; nt < 3; ++nt) acc[mt][nt] = wmma_bf16(af[mt], bf[nt], acc[mt][nt]);
  }
#pragma unroll
  for (int mt = 0; mt < 3; ++mt)
#pragma unroll
    for (int nt = 0; nt < 3; ++nt)
#pragma unroll
      for (int r = 0; r < 8; ++r)
        atomicAdd(&attnS[mt * 16 + r + 8 * (lane >> 4)][nt * 16 + (lane & 15)],
                  acc[mt][nt][r]);
  __syncthreads();
  if (tid < HD) {
    const float* rq = rnorm + b * 2 * CDIM + head * HD;
    const float* rk = rnorm + b * 2 * CDIM + CDIM + head * HD;
    float T = temp[head];
    float sc0 = rq[tid] * T;
    float mx = -3.4e38f;
    float v[HD];
    for (int n = 0; n < HD; ++n) {
      v[n] = attnS[tid][n] * sc0 * rk[n];
      mx = fmaxf(mx, v[n]);
    }
    float sum = 0.f;
    for (int n = 0; n < HD; ++n) {
      v[n] = expf(v[n] - mx);
      sum += v[n];
    }
    float inv = 1.0f / sum;
    float* dst = attnG + (((size_t)b * NHEADS + head) * HD + tid) * 64;
    for (int n = 0; n < HD; ++n) dst[n] = v[n] * inv;
    for (int n = HD; n < 64; ++n) dst[n] = 0.f;   // zero pad K for attn@v WMMA
  }
}

// ------------------------------------------------- K10: out = (attn @ v) * sigmoid(gate)
__global__ void applyattn_kernel(const float* __restrict__ attnG,
                                 const bf16_t* __restrict__ qkvdw,
                                 bf16_t* __restrict__ outg) {
  int n0 = blockIdx.x * 128, head = blockIdx.y, b = blockIdx.z;
  __shared__ alignas(16) bf16_t atile[HD][64];
  __shared__ alignas(16) bf16_t btile[128][64];
  int tid = threadIdx.x, lane = tid & 31, wid = tid >> 5;
  const bf16_t* vbase = qkvdw + ((size_t)b * C4 + 3 * CDIM + head * HD) * NPIX;
  for (int i = tid; i < HD * 64; i += 256) {
    int row = i >> 6, kc = i & 63;
    atile[row][kc] =
        (bf16_t)attnG[(((size_t)b * NHEADS + head) * HD + row) * 64 + kc];
  }
  for (int c = tid; c < 64 * 128 / 8; c += 256) {   // wide read, transposed store
    int k = c >> 4, c8 = c & 15;
    v8bf d;
    if (k < HD) {
      d = *(const v8bf*)(vbase + (size_t)k * NPIX + n0 + c8 * 8);
    } else {
#pragma unroll
      for (int j = 0; j < 8; ++j) d[j] = (bf16_t)0.f;
    }
#pragma unroll
    for (int j = 0; j < 8; ++j) btile[c8 * 8 + j][k] = d[j];
  }
  __syncthreads();
  v8f acc[3] = {};
#pragma unroll
  for (int kk = 0; kk < 64; kk += 32)
#pragma unroll
    for (int mt = 0; mt < 3; ++mt) {
      v16bf af = load_a_frag(&atile[mt * 16 + (lane & 15)][kk], lane >> 4);
      v16bf bf = load_b_frag(&btile[wid * 16 + (lane & 15)][kk], lane >> 4);
      acc[mt] = wmma_bf16(af, bf, acc[mt]);
    }
  const bf16_t* gbase = qkvdw + ((size_t)b * C4 + CDIM + head * HD) * NPIX;
  bf16_t* obase = outg + ((size_t)b * CDIM + head * HD) * NPIX;
#pragma unroll
  for (int mt = 0; mt < 3; ++mt)
#pragma unroll
    for (int r = 0; r < 8; ++r) {
      int m = mt * 16 + r + 8 * (lane >> 4);
      int n = n0 + wid * 16 + (lane & 15);
      float g = (float)gbase[(size_t)m * NPIX + n];
      obase[(size_t)m * NPIX + n] = (bf16_t)(acc[mt][r] * sigmoidf_(g));
    }
}

// ----------------------------------------------------------------- launcher
extern "C" void kernel_launch(void* const* d_in, const int* in_sizes, int n_in,
                              void* d_out, int out_size, void* d_ws, size_t ws_size,
                              hipStream_t stream) {
  (void)in_sizes; (void)n_in; (void)out_size; (void)ws_size;
  const float* x     = (const float*)d_in[0];
  const float* wfc1  = (const float*)d_in[1];
  const float* wfc2  = (const float*)d_in[2];
  const float* wsp   = (const float*)d_in[3];
  const float* wqkvf = (const float*)d_in[4];
  const float* wdw   = (const float*)d_in[5];
  const float* wprjf = (const float*)d_in[6];
  const float* temp  = (const float*)d_in[7];

  char* ws = (char*)d_ws;
  size_t off = 0;
  auto take = [&](size_t bytes) -> char* {
    char* p = ws + off;
    off += (bytes + 255) & ~(size_t)255;
    return p;
  };
  float*  pool     = (float*)take((size_t)BATCH * 2 * CDIM * 4);
  float*  cg       = (float*)take((size_t)BATCH * CDIM * 4);
  float*  spstat   = (float*)take((size_t)BATCH * 2 * NPIX * 4);
  float*  sg       = (float*)take((size_t)BATCH * NPIX * 4);
  bf16_t* wqkv_bf  = (bf16_t*)take((size_t)C4 * CDIM * 2);
  bf16_t* wproj_bf = (bf16_t*)take((size_t)CDIM * CDIM * 2);
  bf16_t* xg       = (bf16_t*)take((size_t)BATCH * CDIM * NPIX * 2); // reused as outg
  bf16_t* qkv      = (bf16_t*)take((size_t)BATCH * C4 * NPIX * 2);
  bf16_t* qkvdw    = (bf16_t*)take((size_t)BATCH * C4 * NPIX * 2);
  float*  rnorm    = (float*)take((size_t)BATCH * 2 * CDIM * 4);
  float*  attnG    = (float*)take((size_t)BATCH * NHEADS * HD * 64 * 4);

  // DFRG gates
  pool_kernel<<<BATCH * CDIM, 256, 0, stream>>>(x, pool);
  gate_kernel<<<BATCH, 192, 0, stream>>>(pool, wfc1, wfc2, cg);
  spstat_kernel<<<BATCH * NPIX / 256, 256, 0, stream>>>(x, spstat);
  spconv_kernel<<<BATCH * NPIX / 256, 256, 0, stream>>>(spstat, wsp, sg);
  cvt_bf16_kernel<<<(C4 * CDIM + 255) / 256, 256, 0, stream>>>(wqkvf, wqkv_bf,
                                                               C4 * CDIM);
  cvt_bf16_kernel<<<(CDIM * CDIM + 255) / 256, 256, 0, stream>>>(wprjf, wproj_bf,
                                                                 CDIM * CDIM);
  applygates_kernel<<<(int)((size_t)BATCH * CDIM * NPIX / (256 * 8)), 256, 0,
                      stream>>>(x, cg, sg, xg);
  // qkv = Wqkv @ xg   (bf16 WMMA, f32 accum, bf16 out)
  gemm_kernel<true><<<dim3(NPIX / 128, C4 / 64, BATCH), 256, 0, stream>>>(
      wqkv_bf, xg, qkv, C4, CDIM, (long)CDIM * NPIX, (long)C4 * NPIX);
  // depthwise 3x3
  dwconv_kernel<<<(int)((size_t)BATCH * C4 * NPIX / 256), 256, 0, stream>>>(qkv, wdw,
                                                                            qkvdw);
  // L2 norms for q,k rows
  norm_kernel<<<BATCH * 2 * CDIM, 256, 0, stream>>>(qkvdw, rnorm);
  // attention matrices (softmaxed, padded K=64)
  attn_kernel<<<dim3(NHEADS, BATCH), 256, 0, stream>>>(qkvdw, rnorm, temp, attnG);
  // out = (attn @ v) * sigmoid(gate)  -> reuse xg buffer as outg
  applyattn_kernel<<<dim3(NPIX / 128, NHEADS, BATCH), 256, 0, stream>>>(attnG, qkvdw,
                                                                        xg);
  // final projection: d_out = Wproj @ outg  (f32 out)
  gemm_kernel<false><<<dim3(NPIX / 128, CDIM / 64, BATCH), 256, 0, stream>>>(
      wproj_bf, xg, d_out, CDIM, CDIM, (long)CDIM * NPIX, (long)CDIM * NPIX);
}
